// ExplicitRelationEncoder_14998025798080
// MI455X (gfx1250) — matrix operand order repaired and checked by reading, not compiled, not run on hardware
//
#include <hip/hip_runtime.h>
#include <hip/hip_bf16.h>
#include <stdint.h>

// ---------------------------------------------------------------------------
// Problem constants (from reference setup_inputs)
// ---------------------------------------------------------------------------
#define BB   256
#define NN   36
#define LL   11
#define FF   1024
#define QQ   1024
#define INF_ 2048          // F + Q
#define HH   16
#define DH   64            // F / H
#define NG   20            // min(NONGT, N)
#define BN   (BB * NN)     // 9216 rows
#define BM   (BB * NG)     // 5120 kv rows
#define NEGV (-9.0e15f)

typedef __attribute__((ext_vector_type(16))) _Float16 v16h;
typedef __attribute__((ext_vector_type(8)))  _Float16 v8h;
typedef __attribute__((ext_vector_type(8)))  float    v8f;

__device__ __forceinline__ void wait_async_zero() {
#if __has_builtin(__builtin_amdgcn_s_wait_asynccnt)
    __builtin_amdgcn_s_wait_asynccnt(0);
#else
    asm volatile("s_wait_asynccnt 0x0" ::: "memory");
#endif
}

// ---------------------------------------------------------------------------
// vcq = concat(v, row_zero ? 0 : q) as f16   (one block per (b,n) row)
// ---------------------------------------------------------------------------
__global__ __launch_bounds__(256)
void build_vcq_kernel(const float* __restrict__ v, const float* __restrict__ q,
                      _Float16* __restrict__ vcq)
{
    const int bn = blockIdx.x;
    const int b  = bn / NN;
    const int t  = threadIdx.x;
    const float* vrow = v + (size_t)bn * FF;

    float s = 0.f;
    for (int j = t; j < FF; j += 256) s += vrow[j];
    __shared__ float red[256];
    red[t] = s;
    __syncthreads();
    for (int st = 128; st > 0; st >>= 1) {
        if (t < st) red[t] += red[t + st];
        __syncthreads();
    }
    const bool zero = (red[0] == 0.f);

    const float* qrow = q + (size_t)b * QQ;
    _Float16* dst = vcq + (size_t)bn * INF_;
    for (int j = t; j < FF; j += 256) {
        dst[j]      = (_Float16)vrow[j];
        dst[FF + j] = zero ? (_Float16)0.f : (_Float16)qrow[j];
    }
}

// ---------------------------------------------------------------------------
// f32 -> f16 elementwise convert
// ---------------------------------------------------------------------------
__global__ __launch_bounds__(256)
void cvt_f32_f16_kernel(const float* __restrict__ src, _Float16* __restrict__ dst, int n)
{
    int i = blockIdx.x * 256 + threadIdx.x;
    if (i < n) dst[i] = (_Float16)src[i];
}

// ---------------------------------------------------------------------------
// gather kv rows: kv_h[b*NG+m] = sf_h[b*NN+m], m < NG
// ---------------------------------------------------------------------------
__global__ __launch_bounds__(256)
void gather_kv_kernel(const _Float16* __restrict__ sf_h, _Float16* __restrict__ kv_h)
{
    const int r = blockIdx.x;             // 0..BM-1
    const int b = r / NG, m = r % NG;
    const _Float16* src = sf_h + ((size_t)b * NN + m) * FF;
    _Float16* dst = kv_h + (size_t)r * FF;
    for (int j = threadIdx.x; j < FF; j += 256) dst[j] = src[j];
}

// ---------------------------------------------------------------------------
// Tiled WMMA GEMM: Y[M,N] = X[M,K] * W[N,K]^T + bias[N]     (f16 in, f32 out)
// Block tile 128x128, 8 waves (2x4), wave tile 64x32 = 4x2 wmma frags, K step 32.
// Double-buffered LDS fed by global_load_async_to_lds_b128 (ASYNCcnt pipeline):
// next tile streams into the other buffer while WMMAs consume the current one.
// M % 128 == 0, N % 128 == 0, K % 32 == 0 (true for all call sites).
// ---------------------------------------------------------------------------
__global__ __launch_bounds__(256)
void gemm_f16_wmma_kernel(const _Float16* __restrict__ X,
                          const _Float16* __restrict__ W,
                          const float*    __restrict__ bias,
                          float*          __restrict__ Y,
                          int M, int N, int K)
{
    // [buf][ A(128x40) | B(128x40) ] halfs, padded stride 40 -> 16B-aligned frags
    __shared__ __align__(16) _Float16 smem[2 * 2 * 128 * 40];   // 40 KB

    const int t     = threadIdx.x;
    const int lane  = t & 31;
    const int wave  = t >> 5;
    const int wm    = wave >> 2;          // 0..1
    const int wn    = wave & 3;           // 0..3
    const int row0  = blockIdx.x * 128;
    const int col0  = blockIdx.y * 128;

    const int ldr   = t >> 2;             // 0..63   (loader row)
    const int ldc   = (t & 3) * 8;        // 0,8,16,24 (loader k-chunk, halfs)
    const int laneM = lane & 15;
    const int kA    = (lane < 16) ? 0 : 8;    // A-frag K base (half index)
    const int kB    = (lane < 16) ? 0 : 8;    // B-frag K base (half index)

    // wave-relative LDS byte address of smem (flat shared addr low 32 bits)
    const uint32_t lds_base = (uint32_t)(uintptr_t)smem;
    const uint32_t TILEB    = 128u * 40u * 2u;     // bytes per A (or B) tile
    const uint32_t BUFB     = 2u * TILEB;          // bytes per double-buffer slot

    const _Float16* gA = X + (size_t)(row0 + ldr) * K + ldc;
    const _Float16* gB = W + (size_t)(col0 + ldr) * K + ldc;
    const uint32_t  lo = (uint32_t)(ldr * 40 + ldc) * 2u;

    auto issue_tile = [&](int buf, int kt) {
        const uint32_t la = lds_base + (uint32_t)buf * BUFB + lo;
        const uint32_t lb = la + TILEB;
        #pragma unroll
        for (int it = 0; it < 2; ++it) {
            const _Float16* pa = gA + (size_t)(it * 64) * K + kt;
            const _Float16* pb = gB + (size_t)(it * 64) * K + kt;
            const uint32_t  dd = (uint32_t)(it * 64 * 40) * 2u;
            asm volatile("global_load_async_to_lds_b128 %0, %1, off"
                         :: "v"(la + dd), "v"(pa) : "memory");
            asm volatile("global_load_async_to_lds_b128 %0, %1, off"
                         :: "v"(lb + dd), "v"(pb) : "memory");
        }
    };

    v8f acc[4][2] = {};

    issue_tile(0, 0);
    wait_async_zero();
    __syncthreads();

    const int nk = K >> 5;
    for (int ik = 0; ik < nk; ++ik) {
        const int buf = ik & 1;
        if (ik + 1 < nk) issue_tile(buf ^ 1, (ik + 1) << 5);

        const _Float16* As = smem + (size_t)buf * (2 * 128 * 40);
        const _Float16* Bs = As + 128 * 40;

        // ---- B fragments: lane n=lane%16; K halves {kB..kB+7, kB+16..kB+23}
        v16h bf[2];
        #pragma unroll
        for (int ni = 0; ni < 2; ++ni) {
            const int n = wn * 32 + ni * 16 + laneM;
            const v8h blo = *(const v8h*)(&Bs[n * 40 + kB]);
            const v8h bhi = *(const v8h*)(&Bs[n * 40 + kB + 16]);
            bf[ni] = __builtin_shufflevector(blo, bhi, 0,1,2,3,4,5,6,7,8,9,10,11,12,13,14,15);
        }

        // ---- A fragments + WMMA: lane m=lane%16; K halves {kA..kA+7, kA+16..kA+23}
        #pragma unroll
        for (int mi = 0; mi < 4; ++mi) {
            const int r = wm * 64 + mi * 16 + laneM;
            const v8h alo = *(const v8h*)(&As[r * 40 + kA]);
            const v8h ahi = *(const v8h*)(&As[r * 40 + kA + 16]);
            const v16h af = __builtin_shufflevector(alo, ahi, 0,1,2,3,4,5,6,7,8,9,10,11,12,13,14,15);
            #pragma unroll
            for (int ni = 0; ni < 2; ++ni)
                acc[mi][ni] = __builtin_amdgcn_wmma_f32_16x16x32_f16(
                    false, af, false, bf[ni], (short)0, acc[mi][ni], false, false);
        }

        wait_async_zero();   // next tile landed in LDS
        __syncthreads();     // all waves' reads of cur + writes of nxt complete
    }

    // ---- epilogue: C layout — VGPR j holds row j + 8*(lane>=16), col = lane%16
    #pragma unroll
    for (int mi = 0; mi < 4; ++mi) {
        const int rbase = row0 + wm * 64 + mi * 16 + (lane >> 4) * 8;
        #pragma unroll
        for (int ni = 0; ni < 2; ++ni) {
            const int col = col0 + wn * 32 + ni * 16 + laneM;
            const float bv = bias[col];
            #pragma unroll
            for (int j = 0; j < 8; ++j)
                Y[(size_t)(rbase + j) * N + col] = acc[mi][ni][j] + bv;
        }
    }
    (void)M;
}

// ---------------------------------------------------------------------------
// Attention epilogue per (b,n): masked scaled-dot softmax over m<20, then
// acc[b,n,h*64+g] += sum_m w[m] * kvW[b,m,h*64+g]   (bout already folded into kvW
// as GEMM bias; valid because softmax weights sum to 1).
// 512 threads = 16 waves, one wave per head.
// ---------------------------------------------------------------------------
__global__ __launch_bounds__(512)
void attn_epilogue_kernel(const int*   __restrict__ adj,
                          const float* __restrict__ w_bias,
                          const float* __restrict__ b_bias,
                          const float* __restrict__ qh,
                          const float* __restrict__ kh,
                          const float* __restrict__ kvw,
                          float*       __restrict__ acc,
                          int d)
{
    const int bn = blockIdx.x;
    const int b  = bn / NN, n = bn % NN;
    const int t  = threadIdx.x;

    __shared__ float s_vb[NG];
    __shared__ int   s_msk[NG];

    if (t < NG) {
        const int m = t;
        float sum = 0.f, vb = 0.f;
        #pragma unroll
        for (int l = 0; l < LL; ++l) {
            const size_t idx = (d == 0)
                ? ((((size_t)b * NN + n) * NN + m) * LL + l)
                : ((((size_t)b * NN + m) * NN + n) * LL + l);
            const float a = (float)adj[idx];
            sum += a;
            vb  += a * w_bias[l];
        }
        s_vb[m]  = vb + b_bias[0];
        s_msk[m] = (sum > 0.f) ? 1 : 0;
    }
    __syncthreads();

    const int h    = t >> 5;      // head = wave id (0..15)
    const int lane = t & 31;

    const float2 qv = *(const float2*)(qh + (size_t)bn * FF + h * DH + lane * 2);

    // affinity: lane m holds aff[m] for m<20
    float myaff = -3.0e38f;
    for (int m = 0; m < NG; ++m) {
        const float2 kv2 = *(const float2*)(kh + ((size_t)b * NG + m) * FF + h * DH + lane * 2);
        float p = qv.x * kv2.x + qv.y * kv2.y;
        #pragma unroll
        for (int off = 16; off > 0; off >>= 1) p += __shfl_xor(p, off, 32);
        if (lane == m)
            myaff = (s_msk[m] ? p * 0.125f : NEGV) + s_vb[m];
    }

    // softmax over lanes 0..19
    float mx = myaff;
    #pragma unroll
    for (int off = 16; off > 0; off >>= 1) mx = fmaxf(mx, __shfl_xor(mx, off, 32));
    const float e = (lane < NG) ? __expf(myaff - mx) : 0.f;
    float se = e;
    #pragma unroll
    for (int off = 16; off > 0; off >>= 1) se += __shfl_xor(se, off, 32);
    const float wgt = e / se;

    // weighted combine of kvW rows (bout bias already inside kvW)
    const int g0 = lane * 2;
    float o0 = 0.f, o1 = 0.f;
    for (int m = 0; m < NG; ++m) {
        const float wm = __shfl(wgt, m, 32);
        const float2 kw = *(const float2*)(kvw + ((size_t)b * NG + m) * FF + h * DH + g0);
        o0 += wm * kw.x;
        o1 += wm * kw.y;
    }
    float* arow = acc + (size_t)bn * FF + h * DH + g0;
    arow[0] += o0;
    arow[1] += o1;
}

// ---------------------------------------------------------------------------
// out = v + relu(acc)
// ---------------------------------------------------------------------------
__global__ __launch_bounds__(256)
void residual_relu_kernel(const float* __restrict__ v, const float* __restrict__ acc,
                          float* __restrict__ out, int n)
{
    int i = blockIdx.x * 256 + threadIdx.x;
    if (i < n) out[i] = v[i] + fmaxf(acc[i], 0.f);
}

// ---------------------------------------------------------------------------
// Host launch
// ---------------------------------------------------------------------------
extern "C" void kernel_launch(void* const* d_in, const int* in_sizes, int n_in,
                              void* d_out, int out_size, void* d_ws, size_t ws_size,
                              hipStream_t stream)
{
    const float* v      = (const float*)d_in[0];
    const float* q      = (const float*)d_in[1];
    const int*   adj    = (const int*)  d_in[2];
    const float* W_self = (const float*)d_in[3];
    const float* b_self = (const float*)d_in[4];
    const float* w_bias = (const float*)d_in[5];
    const float* b_bias = (const float*)d_in[6];
    const float* Wq     = (const float*)d_in[7];
    const float* bq     = (const float*)d_in[8];
    const float* Wk     = (const float*)d_in[9];
    const float* bk     = (const float*)d_in[10];
    const float* Wout   = (const float*)d_in[11];
    const float* bout   = (const float*)d_in[12];
    float* out = (float*)d_out;

    // ---- workspace carve-up (all 256B aligned) ----
    char* p = (char*)d_ws;
    auto carve = [&](size_t bytes) -> void* {
        void* r = (void*)p;
        p += (bytes + 255) & ~(size_t)255;
        return r;
    };
    _Float16* Wself_h = (_Float16*)carve((size_t)INF_ * FF * 2);
    _Float16* Wq_h    = (_Float16*)carve((size_t)2 * FF * FF * 2);
    _Float16* Wk_h    = (_Float16*)carve((size_t)2 * FF * FF * 2);
    _Float16* Wout_h  = (_Float16*)carve((size_t)2 * FF * FF * 2);
    _Float16* vcq_h   = (_Float16*)carve((size_t)BN * INF_ * 2);
    float*    acc     = (float*)   carve((size_t)BN * FF * 4);   // self_feat, then += att
    _Float16* sf_h    = (_Float16*)carve((size_t)BN * FF * 2);
    _Float16* kv_h    = (_Float16*)carve((size_t)BM * FF * 2);
    float*    kh_f    = (float*)   carve((size_t)BM * FF * 4);
    float*    kvw_f   = (float*)   carve((size_t)BM * FF * 4);
    float*    qh_f    = (float*)vcq_h;   // alias: vcq dead after first GEMM (same size)

    // ---- convert weights f32 -> f16 ----
    {
        const int nws = INF_ * FF;
        cvt_f32_f16_kernel<<<(nws + 255) / 256, 256, 0, stream>>>(W_self, Wself_h, nws);
        const int nw = 2 * FF * FF;
        cvt_f32_f16_kernel<<<(nw + 255) / 256, 256, 0, stream>>>(Wq, Wq_h, nw);
        cvt_f32_f16_kernel<<<(nw + 255) / 256, 256, 0, stream>>>(Wk, Wk_h, nw);
        cvt_f32_f16_kernel<<<(nw + 255) / 256, 256, 0, stream>>>(Wout, Wout_h, nw);
    }

    // ---- vcq = concat(v, row_zero ? 0 : q) in f16 ----
    build_vcq_kernel<<<BN, 256, 0, stream>>>(v, q, vcq_h);

    // ---- self_feat = vcq @ W_self^T + b_self ----
    gemm_f16_wmma_kernel<<<dim3(BN / 128, FF / 128), 256, 0, stream>>>(
        vcq_h, Wself_h, b_self, acc, BN, FF, INF_);

    // ---- f16 copy of self_feat + kv gather ----
    cvt_f32_f16_kernel<<<(BN * FF + 255) / 256, 256, 0, stream>>>(acc, sf_h, BN * FF);
    gather_kv_kernel<<<BM, 256, 0, stream>>>(sf_h, kv_h);

    // ---- two directional GAT passes ----
    for (int d = 0; d < 2; ++d) {
        const size_t woff = (size_t)d * FF * FF;
        gemm_f16_wmma_kernel<<<dim3(BN / 128, FF / 128), 256, 0, stream>>>(
            sf_h, Wq_h + woff, bq + d * FF, qh_f, BN, FF, FF);
        gemm_f16_wmma_kernel<<<dim3(BM / 128, FF / 128), 256, 0, stream>>>(
            kv_h, Wk_h + woff, bk + d * FF, kh_f, BM, FF, FF);
        // kvW = kv @ Wout^T + bout  (bout folded here; softmax weights sum to 1)
        gemm_f16_wmma_kernel<<<dim3(BM / 128, FF / 128), 256, 0, stream>>>(
            kv_h, Wout_h + woff, bout + d * FF, kvw_f, BM, FF, FF);

        attn_epilogue_kernel<<<BN, 512, 0, stream>>>(
            adj, w_bias, b_bias, qh_f, kh_f, kvw_f, acc, d);
    }

    // ---- out = v + relu(self_feat + att0 + att1) ----
    residual_relu_kernel<<<(BN * FF + 255) / 256, 256, 0, stream>>>(v, acc, out, BN * FF);

    (void)in_sizes; (void)n_in; (void)out_size; (void)ws_size;
}
